// LG_cross_44461501448910
// MI455X (gfx1250) — matrix operand order repaired and checked by reading, not compiled, not run on hardware
//
#include <hip/hip_runtime.h>

// ---------------------------------------------------------------------------
// GCN: embed = adj@((adj@(x@W2)+b2) @ W4)+b4 + (adj@(x@W2)+b2)@res_w^T + res_b
// N = 16384, all feature dims = 128.
// Dominant cost: two adj[16384x16384] @ support[16384x128] GEMMs (~69 GFLOP ea).
// HBM plan: stream adj as f32 (2 GiB total -> ~86us floor @ 23.3 TB/s),
//   convert to bf16 in-register (cheaper than a separate bf16 pre-pass).
// L2 plan: supports (4 MB bf16) staged block-cooperatively through LDS
//   (double-buffered, barrier-pipelined): each block reads them from L2 once.
// Compute: v_wmma_f32_16x16x32_bf16, f32 accumulation. Scheduling-group
//   hints batch all 16 ds_load_b128 ahead of the 8 WMMAs per K-step so the
//   wave drains matrix ops back-to-back instead of load->wait0->wmma chains.
// ---------------------------------------------------------------------------

typedef __attribute__((ext_vector_type(16))) __bf16       v16bf;
typedef __attribute__((ext_vector_type(8)))  float        v8f;
typedef __attribute__((ext_vector_type(4)))  float        v4f;
typedef __attribute__((ext_vector_type(4)))  unsigned int v4u;

#define NN   16384   // graph size (rows of adj, K of aggregation)
#define DF   128     // feature dim (in/hidden/out all 128)

// sched_group_barrier masks (AMDGPU): 0x002 VALU, 0x008 MFMA/WMMA,
// 0x020 VMEM read, 0x100 DS read.
#define SGB(mask, n) __builtin_amdgcn_sched_group_barrier((mask), (n), 0)

// ---------------------------------------------------------------------------
// Projection 1: ST[j][i] = bf16( sum_k x[i][k] * W[k][j] )   (support^T)
// ---------------------------------------------------------------------------
__global__ __launch_bounds__(128)
void proj_xW_kernel(const float* __restrict__ X, const float* __restrict__ W,
                    __bf16* __restrict__ STout)
{
    __shared__ float xs[DF];
    const int i = blockIdx.x;
    const int j = threadIdx.x;
    xs[j] = X[(size_t)i * DF + j];
    __syncthreads();
    float s = 0.f;
#pragma unroll 8
    for (int k = 0; k < DF; ++k)
        s += xs[k] * W[k * DF + j];               // coalesced over j
    STout[(size_t)j * NN + i] = (__bf16)s;        // transposed bf16 store
}

// ---------------------------------------------------------------------------
// Projection 2 (fused): ST2 = (H @ W4)^T bf16 ; Out = H @ res_w^T + res_b
// ---------------------------------------------------------------------------
__global__ __launch_bounds__(128)
void proj_hidden_kernel(const float* __restrict__ H, const float* __restrict__ W4,
                        const float* __restrict__ RW, const float* __restrict__ RB,
                        __bf16* __restrict__ ST2, float* __restrict__ Out)
{
    __shared__ float hs[DF];
    const int i = blockIdx.x;
    const int j = threadIdx.x;
    hs[j] = H[(size_t)i * DF + j];
    __syncthreads();
    float s = 0.f, r = 0.f;
#pragma unroll 4
    for (int k = 0; k < DF; ++k) {
        const float h = hs[k];
        s += h * W4[k * DF + j];
        r += h * RW[j * DF + k];                  // res_w[j][k] (64KB, cached)
    }
    ST2[(size_t)j * NN + i] = (__bf16)s;
    Out[(size_t)i * DF + j] = r + RB[j];
}

// ---------------------------------------------------------------------------
// Aggregation: out[m][n] = sum_k adj[m][k] * ST[n][k] + bias[n] (+ accum)
// Block = 256 threads = 8 waves; each wave owns a 16-row x 128-col strip
// (adj read exactly once from HBM). Support staged through LDS:
//   chunk = 64 K-steps x 128 cols bf16 = 16 KB, double-buffered (36 KB LDS).
//   LDS col stride = 144 B (128 data + 16 pad): stride/4 = 36 = 4*9 (9 odd)
//   -> the 16 cols of a half-wave tile start on distinct banks.
// WMMA register layouts (CDNA5 ISA, 16-bit A 16x32 / B 32x16):
//   A: lane&15 = row, hi=lane>>4 -> K chunks [hi*8,+8) and [hi*8+16,+8)
//   B: lane&15 = col, hi -> K chunks [hi*8,+8) and [hi*8+16,+8)
//   C: VGPR v -> M = hi*8 + v, N = lane&15
// ---------------------------------------------------------------------------
__global__ __launch_bounds__(256)
void gcn_aggregate_kernel(const float* __restrict__ adj,
                          const __bf16* __restrict__ ST,
                          const float* __restrict__ bias,
                          const float* __restrict__ accum,
                          float* __restrict__ out,
                          int addAccum)
{
    constexpr int KC     = 64;    // K per LDS chunk (2 WMMA K-steps)
    constexpr int STRIDE = 144;   // LDS bytes per support column (pad vs banks)
    constexpr int NCHUNK = NN / KC;
    __shared__ __align__(16) unsigned char smem[2][DF * STRIDE];  // 36 KB

    const int tid   = threadIdx.x;
    const int lane  = tid & 31;
    const int wave  = tid >> 5;
    const int rowbase = (blockIdx.x * 8 + wave) * 16;
    const int mrow  = lane & 15;       // A row / B col within a tile
    const int hi    = lane >> 4;       // lane-group select
    const int akoff = hi * 8;          // A chunk K offset
    const int bkoff = hi * 8;          // B chunk K offset

    const float* aptr = adj + (size_t)(rowbase + mrow) * NN;

    v8f acc[8];
#pragma unroll
    for (int t = 0; t < 8; ++t) acc[t] = v8f{};

    // Cooperative staging map: 1024 x 16B units per chunk, 4 per thread.
    // unit -> col = unit>>3, 16B-slot = unit&7 (consecutive tids walk K: coalesced).
    int ucol[4], uelt[4], ubyte[4];
#pragma unroll
    for (int r = 0; r < 4; ++r) {
        const int unit = tid + 256 * r;
        ucol[r]  = unit >> 3;
        uelt[r]  = (unit & 7) * 8;     // bf16 elements into the chunk
        ubyte[r] = (unit & 7) * 16;    // bytes into the LDS column
    }

    // Preload chunk 0 into buffer 0.
    v4u sreg[4];
#pragma unroll
    for (int r = 0; r < 4; ++r)
        sreg[r] = *(const v4u*)(ST + (size_t)ucol[r] * NN + uelt[r]);
#pragma unroll
    for (int r = 0; r < 4; ++r)
        *(v4u*)(&smem[0][ucol[r] * STRIDE + ubyte[r]]) = sreg[r];
    __syncthreads();

    for (int c = 0; c < NCHUNK; ++c) {
        const int k0 = c * KC;
        const int cb = c & 1;
        const bool more = (c + 1) < NCHUNK;

        // Kick off global loads for the NEXT chunk (overlap with compute).
        if (more) {
#pragma unroll
            for (int r = 0; r < 4; ++r)
                sreg[r] = *(const v4u*)(ST + (size_t)ucol[r] * NN + (k0 + KC) + uelt[r]);
        }
        // Prefetch the streaming adj rows ~2KB ahead (global_prefetch_b8).
        __builtin_prefetch((const void*)(aptr + k0 + 512), 0, 0);

        // ---- compute 2 K-steps out of LDS buffer cb ----
#pragma unroll
        for (int kk = 0; kk < KC; kk += 32) {
            // A tile: 16 f32 -> 16 bf16 in WMMA register layout
            const float* ak = aptr + k0 + kk + akoff;
            v4f a0 = *(const v4f*)(ak);
            v4f a1 = *(const v4f*)(ak + 4);
            v4f a2 = *(const v4f*)(ak + 16);
            v4f a3 = *(const v4f*)(ak + 20);
            v16bf a;
#pragma unroll
            for (int e = 0; e < 4; ++e) {
                a[e]      = (__bf16)a0[e];   // K = akoff + 0..3
                a[4 + e]  = (__bf16)a1[e];   // K = akoff + 4..7
                a[8 + e]  = (__bf16)a2[e];   // K = akoff + 16..19
                a[12 + e] = (__bf16)a3[e];   // K = akoff + 20..23
            }

            // B tiles from LDS (two ds_load_b128 each; 16B-aligned)
            v16bf b[8];
#pragma unroll
            for (int t = 0; t < 8; ++t) {
                const int off = (t * 16 + mrow) * STRIDE + (kk + bkoff) * 2;
                union { v4u u[2]; v16bf v; } cv;
                cv.u[0] = *(const v4u*)(&smem[cb][off]);
                cv.u[1] = *(const v4u*)(&smem[cb][off + 32]);
                b[t] = cv.v;
            }

            // 8 x v_wmma_f32_16x16x32_bf16
#pragma unroll
            for (int t = 0; t < 8; ++t)
                acc[t] = __builtin_amdgcn_wmma_f32_16x16x32_bf16(
                    false, a, false, b[t], (short)0, acc[t], false, false);

            // Issue-order request for this K-step:
            //   A vmem loads -> cvt VALU -> ALL 16 ds reads -> 8 WMMAs.
            // Batching the DS reads ahead of the WMMAs turns the
            // load->wait(0)->wmma chain into a pipelined drain.
            SGB(0x020, 4);    // 4 global_load_b128 (A tile)
            SGB(0x002, 8);    // 8 v_cvt_pk_bf16_f32
            SGB(0x100, 16);   // 16 ds_load_b128 (all B tiles)
            SGB(0x008, 8);    // 8 v_wmma
        }

        // Commit the prefetched chunk into the other LDS buffer.
        if (more) {
#pragma unroll
            for (int r = 0; r < 4; ++r)
                *(v4u*)(&smem[cb ^ 1][ucol[r] * STRIDE + ubyte[r]]) = sreg[r];
        }
        __syncthreads();
    }

    // ---- Epilogue: bias (+ optional accumulate), f32 store ----
    const int orow0 = rowbase + hi * 8;
#pragma unroll
    for (int t = 0; t < 8; ++t) {
        const int col = t * 16 + mrow;
        const float bv = bias[col];
#pragma unroll
        for (int v = 0; v < 8; ++v) {
            const size_t idx = (size_t)(orow0 + v) * DF + col;
            float r = acc[t][v] + bv;
            if (addAccum) r += accum[idx];
            out[idx] = r;
        }
    }
}

// ---------------------------------------------------------------------------
// Launch. Inputs in reference dict order:
//   0:x[N,128] 1:adj[N,N] 2:W2 3:b2 4:W4 5:b4 6:res_w 7:res_b
// ws layout: ST1 bf16 4MB | H f32 8MB | ST2 bf16 4MB (16MB total)
// ---------------------------------------------------------------------------
extern "C" void kernel_launch(void* const* d_in, const int* in_sizes, int n_in,
                              void* d_out, int out_size, void* d_ws, size_t ws_size,
                              hipStream_t stream)
{
    const float* x     = (const float*)d_in[0];
    const float* adj   = (const float*)d_in[1];
    const float* W2    = (const float*)d_in[2];
    const float* b2    = (const float*)d_in[3];
    const float* W4    = (const float*)d_in[4];
    const float* b4    = (const float*)d_in[5];
    const float* res_w = (const float*)d_in[6];
    const float* res_b = (const float*)d_in[7];
    float* out = (float*)d_out;

    char* ws = (char*)d_ws;
    __bf16* ST1 = (__bf16*)(ws);                                   // 4MB
    float*  H   = (float*)(ws + (size_t)DF * NN * sizeof(__bf16)); // 8MB
    __bf16* ST2 = (__bf16*)(ws + (size_t)DF * NN * sizeof(__bf16)
                               + (size_t)NN * DF * sizeof(float)); // 4MB

    // 1) support1^T = (x @ W2)^T  (bf16)
    proj_xW_kernel<<<NN, 128, 0, stream>>>(x, W2, ST1);

    // 2) H = adj @ support1 + b2  (f32 hidden)
    gcn_aggregate_kernel<<<NN / (16 * 8), 256, 0, stream>>>(
        adj, ST1, b2, out, H, 0);

    // 3) support2^T = (H @ W4)^T (bf16); out = H @ res_w^T + res_b (residual)
    proj_hidden_kernel<<<NN, 128, 0, stream>>>(H, W4, res_w, res_b, ST2, out);

    // 4) out += adj @ support2 + b4
    gcn_aggregate_kernel<<<NN / (16 * 8), 256, 0, stream>>>(
        adj, ST2, b4, out, out, 1);
}
